// SensitivitySpecificityLoss_9139690406389
// MI455X (gfx1250) — 8-bit matrix operand order repaired and checked by reading, compiled, not run on hardware
//
#include <hip/hip_runtime.h>
#include <stdint.h>

typedef __attribute__((ext_vector_type(8))) int v8i;

#define C_CLS 21
#define HW_IMG (512 * 512)
#define NPIX (16 * HW_IMG)
#define NCHUNK (NPIX / 64)
#define THREADS 256
#define BLOCKS 256
#define WAVES_PER_BLOCK (THREADS / 32)

__device__ __forceinline__ uint32_t byte_onehot(uint32_t w, uint32_t m) {
    uint32_t r = 0u;
    r |= (((w      ) & 0xFFu) == m) ? 0x00000001u : 0u;
    r |= (((w >>  8) & 0xFFu) == m) ? 0x00000100u : 0u;
    r |= (((w >> 16) & 0xFFu) == m) ? 0x00010000u : 0u;
    r |= (((w >> 24)        ) == m) ? 0x01000000u : 0u;
    return r;
}

__global__ void __launch_bounds__(512) zero_conf_kernel(int* __restrict__ g_conf) {
    int i = threadIdx.x;
    if (i < C_CLS * C_CLS) g_conf[i] = 0;
}

__global__ void __launch_bounds__(THREADS)
conf_wmma_kernel(const float* __restrict__ logits,
                 const int*   __restrict__ target,
                 int*         __restrict__ g_conf) {
    __shared__ uint8_t sm_t[WAVES_PER_BLOCK][64];
    __shared__ uint8_t sm_p[WAVES_PER_BLOCK][64];
    __shared__ int conf_s[32 * 32];

    const int tid     = threadIdx.x;
    const int lane    = tid & 31;
    const int wib     = tid >> 5;
    const int waveId  = blockIdx.x * WAVES_PER_BLOCK + wib;
    const int nWaves  = gridDim.x * WAVES_PER_BLOCK;
    const int halfsel = lane >> 4;
    const uint32_t m_lo = (uint32_t)(lane & 15);

    for (int i = tid; i < 32 * 32; i += THREADS) conf_s[i] = 0;
    __syncthreads();

    v8i acc00 = {}; v8i acc01 = {}; v8i acc10 = {}; v8i acc11 = {};

    for (int chunk = waveId; chunk < NCHUNK; chunk += nWaves) {
        const int k0 = lane * 2;
        const int p0 = chunk * 64 + k0;
        const int b  = p0 >> 18;
        const int hw = p0 & (HW_IMG - 1);

        const float* base = logits + (size_t)b * (size_t)(C_CLS * HW_IMG) + (size_t)hw;
        float2 v0 = *reinterpret_cast<const float2*>(base);
        float best0 = v0.x, best1 = v0.y;
        int   i0 = 0,      i1 = 0;
#pragma unroll
        for (int c = 1; c < C_CLS; ++c) {
            float2 v = *reinterpret_cast<const float2*>(base + (size_t)c * HW_IMG);
            if (v.x > best0) { best0 = v.x; i0 = c; }
            if (v.y > best1) { best1 = v.y; i1 = c; }
        }

        const int2 t2 = *reinterpret_cast<const int2*>(target + p0);

        *reinterpret_cast<uint16_t*>(&sm_t[wib][k0]) =
            (uint16_t)((t2.x & 0xFF) | ((t2.y & 0xFF) << 8));
        *reinterpret_cast<uint16_t*>(&sm_p[wib][k0]) =
            (uint16_t)((i0 & 0xFF) | ((i1 & 0xFF) << 8));
        __syncthreads();

        v8i A0, A1, B0, B1;
#pragma unroll
        for (int v = 0; v < 8; ++v) {
            const int ka = ((v & 1) << 2) | ((v & 2) << 3) | ((v >> 2) << 5) | (halfsel << 3);
            const uint32_t wt = *reinterpret_cast<const uint32_t*>(&sm_t[wib][ka]);
            A0[v] = (int)byte_onehot(wt, m_lo);
            A1[v] = (int)byte_onehot(wt, m_lo + 16u);
            const int kb = ((v & 1) << 2) | ((v & 2) << 3) | ((v >> 2) << 5) | (halfsel << 3);
            const uint32_t wp = *reinterpret_cast<const uint32_t*>(&sm_p[wib][kb]);
            B0[v] = (int)byte_onehot(wp, m_lo);
            B1[v] = (int)byte_onehot(wp, m_lo + 16u);
        }
        __syncthreads();

        acc00 = __builtin_amdgcn_wmma_i32_16x16x64_iu8(false, A0, false, B0, acc00, false, false);
        acc01 = __builtin_amdgcn_wmma_i32_16x16x64_iu8(false, A0, false, B1, acc01, false, false);
        acc10 = __builtin_amdgcn_wmma_i32_16x16x64_iu8(false, A1, false, B0, acc10, false, false);
        acc11 = __builtin_amdgcn_wmma_i32_16x16x64_iu8(false, A1, false, B1, acc11, false, false);
    }

#pragma unroll
    for (int j = 0; j < 8; ++j) {
        const int r = j + (halfsel << 3);
        const int c = (int)m_lo;
        atomicAdd(&conf_s[(r     ) * 32 + c     ], acc00[j]);
        atomicAdd(&conf_s[(r     ) * 32 + c + 16], acc01[j]);
        atomicAdd(&conf_s[(r + 16) * 32 + c     ], acc10[j]);
        atomicAdd(&conf_s[(r + 16) * 32 + c + 16], acc11[j]);
    }
    __syncthreads();

    for (int i = tid; i < C_CLS * C_CLS; i += THREADS) {
        const int row = i / C_CLS;
        const int col = i - row * C_CLS;
        const int v = conf_s[row * 32 + col];
        if (v != 0) atomicAdd(&g_conf[i], v);
    }
}

__global__ void __launch_bounds__(32)
finalize_kernel(const int* __restrict__ g_conf, float* __restrict__ out) {
    __shared__ float sh[32];
    const int i = threadIdx.x;
    float rowsum = 0.f, colsum = 0.f, tp = 0.f;
    if (i < C_CLS) {
        for (int j = 0; j < C_CLS; ++j) {
            rowsum += (float)g_conf[i * C_CLS + j];
            colsum += (float)g_conf[j * C_CLS + i];
        }
        tp = (float)g_conf[i * C_CLS + i];
    }
    sh[i] = rowsum;
    __syncthreads();
    float total = 0.f;
    for (int j = 0; j < C_CLS; ++j) total += sh[j];
    __syncthreads();
    float metric = 0.f;
    if (i < C_CLS) {
        const float fp = rowsum - tp;
        const float fn = colsum - tp;
        const float tn = total - tp - fp - fn;
        const float smooth = 1.0f;
        const float sens = (tp + smooth) / (tp + fn + smooth);
        const float spec = (tn + smooth) / (tn + fp + smooth);
        metric = 0.5f * sens + 0.5f * spec;
    }
    sh[i] = metric;
    __syncthreads();
    if (i == 0) {
        float s = 0.f;
        for (int j = 0; j < C_CLS; ++j) s += sh[j];
        out[0] = 1.0f - s / (float)C_CLS;
    }
}

extern "C" void kernel_launch(void* const* d_in, const int* in_sizes, int n_in,
                              void* d_out, int out_size, void* d_ws, size_t ws_size,
                              hipStream_t stream) {
    const float* logits = (const float*)d_in[0];
    const int*   target = (const int*)d_in[1];
    int* g_conf = (int*)d_ws;
    float* out  = (float*)d_out;

    zero_conf_kernel<<<1, 512, 0, stream>>>(g_conf);
    conf_wmma_kernel<<<BLOCKS, THREADS, 0, stream>>>(logits, target, g_conf);
    finalize_kernel<<<1, 32, 0, stream>>>(g_conf, out);
}
